// Multi_Head_Attention_54305566491158
// MI455X (gfx1250) — matrix operand order repaired and checked by reading, not compiled, not run on hardware
//
#include <hip/hip_runtime.h>
#include <hip/hip_bf16.h>

// ---------------- types ----------------
typedef __bf16 bf16_t;
typedef __attribute__((ext_vector_type(16))) __bf16 v16bf;
typedef __attribute__((ext_vector_type(8)))  __bf16 v8bf;
typedef __attribute__((ext_vector_type(8)))  float  v8f;
typedef __attribute__((ext_vector_type(4)))  float  v4f;

#define WMMA_BF16(a, b, c) \
  __builtin_amdgcn_wmma_f32_16x16x32_bf16(false, (a), false, (b), (short)0, (c), false, false)

// Problem constants
#define BB 256
#define TT 256
#define CC 384
#define HH 6
#define DD 64

__device__ __forceinline__ v8f zero8() {
  v8f z;
#pragma unroll
  for (int i = 0; i < 8; i++) z[i] = 0.0f;
  return z;
}

// A-fragment (16x32 bf16, M x K): lane holds row M = lane&15.
// halves [0..7]  = K = kbase + (lane>=16 ? 16 : 0) + {0..7}
// halves [8..15] = same + 8
// caller passes p0 = row_base + kbase + ((lane>>4)<<4)
__device__ __forceinline__ v16bf load_a_bf16(const bf16_t* p0) {
  v8bf lo = *(const v8bf*)(p0);
  v8bf hi = *(const v8bf*)(p0 + 8);
  v16bf r;
#pragma unroll
  for (int i = 0; i < 8; i++) { r[i] = lo[i]; r[8 + i] = hi[i]; }
  return r;
}

// ---------------- kernel 1: pack weights into B-fragment layout ----------------
// B-fragment (32x16 bf16, K x N): lane l holds N = l&15,
// contraction values K = kt*32 + ((l>=16)?16:0) + {0..15} packed 2/VGPR.
// packed[frag*512 + lane*16 + i] = W[K = kbase+i, N]
__global__ __launch_bounds__(32) void pack_weights(
    const float* __restrict__ Wq, const float* __restrict__ Wk,
    const float* __restrict__ Wv, const float* __restrict__ Wp,
    bf16_t* __restrict__ packedQKV, bf16_t* __restrict__ packedP) {
  int f = blockIdx.x;
  int lane = threadIdx.x;
  int n_local = lane & 15;
  int kbase_lane = (lane >> 4) << 4;
  if (f < 864) {  // 3 mats * 6 heads * 12 kt * 4 nt
    int mat = f / 288, rem = f % 288;
    int h = rem / 48, r2 = rem % 48;
    int kt = r2 / 4, nt = r2 % 4;
    const float* W = (mat == 0) ? Wq : ((mat == 1) ? Wk : Wv);
    int n = nt * 16 + n_local;
    int kb = kt * 32 + kbase_lane;
    bf16_t* dst = packedQKV + (size_t)f * 512 + lane * 16;
#pragma unroll
    for (int i = 0; i < 16; i++)
      dst[i] = (bf16_t)W[((size_t)h * CC + (kb + i)) * DD + n];
  } else {        // Wproj: 12 kt * 24 nt
    int g = f - 864;
    int kt = g / 24, nt = g % 24;
    int n = nt * 16 + n_local;
    int kb = kt * 32 + kbase_lane;
    bf16_t* dst = packedP + (size_t)g * 512 + lane * 16;
#pragma unroll
    for (int i = 0; i < 16; i++)
      dst[i] = (bf16_t)Wp[(size_t)(kb + i) * CC + n];
  }
}

// ---------------- kernel 2: fused QKV projection ----------------
// grid.x = 4096 row tiles (16 rows of B*T each); 576 threads = 18 waves,
// wave w = mat*6 + h computes a 16x64 tile of Q/K/V for head h.
// Q,K stored [b,h,t,d] row-major bf16; V stored transposed [b,h,d,t] bf16.
__global__ __launch_bounds__(576) void qkv_gemm(
    const float* __restrict__ x, const bf16_t* __restrict__ packedW,
    bf16_t* __restrict__ Qw, bf16_t* __restrict__ Kw, bf16_t* __restrict__ Vt) {
  __shared__ bf16_t xb[16 * CC];          // 12 KB staged x tile (bf16)
  __shared__ bf16_t obuf[12 * 16 * 64];   // 24 KB transpose staging for Q/K waves
  int tid = threadIdx.x;
  int m0 = blockIdx.x * 16;

  // Stage x tile once: 6144 floats -> bf16 in LDS
  for (int i = tid * 4; i < 16 * CC; i += 576 * 4) {
    v4f v = *(const v4f*)(x + (size_t)m0 * CC + i);
    bf16_t* d = xb + i;
    d[0] = (bf16_t)v[0]; d[1] = (bf16_t)v[1];
    d[2] = (bf16_t)v[2]; d[3] = (bf16_t)v[3];
  }
  __syncthreads();

  int wave = tid >> 5, lane = tid & 31;
  int mat = wave / 6, h = wave % 6;
  int col = lane & 15, rofs = (lane >> 4) << 3;

  const bf16_t* arow = xb + (size_t)col * CC + ((lane >> 4) << 4);
  v8f acc[4];
#pragma unroll
  for (int nt = 0; nt < 4; nt++) acc[nt] = zero8();

  size_t fbase = (size_t)wave * 48;  // ((mat*6+h)*12)*4
#pragma unroll
  for (int kt = 0; kt < 12; kt++) {
    v16bf a = load_a_bf16(arow + kt * 32);
#pragma unroll
    for (int nt = 0; nt < 4; nt++) {
      v16bf bfrag = *(const v16bf*)(packedW + (fbase + kt * 4 + nt) * 512 + lane * 16);
      acc[nt] = WMMA_BF16(a, bfrag, acc[nt]);
    }
  }

  int b = blockIdx.x >> 4;             // 16 tiles per sequence (T=256)
  int t0 = (blockIdx.x & 15) << 4;
  size_t bh = (size_t)b * HH + h;

  if (mat == 2) {
    // V: C-layout lane has fixed d = nt*16+col, rows t contiguous over r -> one b128 store
#pragma unroll
    for (int nt = 0; nt < 4; nt++) {
      int d = nt * 16 + col;
      v8bf o8;
#pragma unroll
      for (int r = 0; r < 8; r++) o8[r] = (bf16_t)acc[nt][r];
      *(v8bf*)(Vt + bh * (DD * TT) + (size_t)d * TT + t0 + rofs) = o8;
    }
  }
  __syncthreads();
  if (mat < 2) {
    bf16_t* ob = obuf + wave * 1024;
#pragma unroll
    for (int nt = 0; nt < 4; nt++)
#pragma unroll
      for (int r = 0; r < 8; r++)
        ob[(r + rofs) * 64 + nt * 16 + col] = (bf16_t)acc[nt][r];
  }
  __syncthreads();
  if (mat < 2) {
    const bf16_t* ob = obuf + wave * 1024;
    bf16_t* base = (mat == 0) ? Qw : Kw;
    int row = lane >> 1, cb = (lane & 1) * 32;
    bf16_t* dst = base + bh * (TT * DD) + (size_t)(t0 + row) * DD + cb;
    const bf16_t* src = ob + row * 64 + cb;
#pragma unroll
    for (int i = 0; i < 4; i++) *(v8bf*)(dst + i * 8) = *(const v8bf*)(src + i * 8);
  }
}

// ---------------- kernel 3: causal attention per (b,h, 16-row tile) ----------------
// Single wave per block (barriers are S_NOP). Scores held in C-fragment regs.
__global__ __launch_bounds__(32) void attention(
    const bf16_t* __restrict__ Qw, const bf16_t* __restrict__ Kw,
    const bf16_t* __restrict__ Vt, bf16_t* __restrict__ aO) {
  __shared__ bf16_t pbuf[16 * 32];   // P transpose bounce (1 KB)
  __shared__ bf16_t obuf[16 * 64];   // output staging (2 KB)

  int lane = threadIdx.x;
  int mt = blockIdx.x & 15;
  int bh = blockIdx.x >> 4;          // [0, 1536)
  const bf16_t* q  = Qw + (size_t)bh * (TT * DD);
  const bf16_t* k  = Kw + (size_t)bh * (TT * DD);
  const bf16_t* vt = Vt + (size_t)bh * (DD * TT);
  int m0 = mt * 16;
  int col = lane & 15, rofs = (lane >> 4) << 3;

  // Q A-fragments (d = 0..31 and 32..63)
  const bf16_t* qrow = q + (size_t)(m0 + col) * DD + ((lane >> 4) << 4);
  v16bf qa0 = load_a_bf16(qrow);
  v16bf qa1 = load_a_bf16(qrow + 32);

  // scores: up to 16 key-tiles, each a 16x16 f32 C-fragment
  v8f sc[16];
#pragma unroll
  for (int kt = 0; kt < 16; kt++) {
    if (kt <= mt) {
      const bf16_t* krow = k + (size_t)(kt * 16 + col) * DD + ((lane >> 4) << 4);
      v16bf kb0 = *(const v16bf*)(krow);
      v16bf kb1 = *(const v16bf*)(krow + 32);
      v8f c = zero8();
      c = WMMA_BF16(qa0, kb0, c);
      c = WMMA_BF16(qa1, kb1, c);
      sc[kt] = c;
    }
  }

  // scale + causal mask + row max (rows live in fixed VGPR index per 16-lane half)
  const float scale = 0.125f;  // 1/sqrt(64)
  float rmax[8];
#pragma unroll
  for (int r = 0; r < 8; r++) rmax[r] = -__builtin_inff();
#pragma unroll
  for (int kt = 0; kt < 16; kt++) {
    if (kt <= mt) {
#pragma unroll
      for (int r = 0; r < 8; r++) {
        float v = sc[kt][r] * scale;
        bool ok = (kt * 16 + col) <= (m0 + r + rofs);
        v = ok ? v : -__builtin_inff();
        sc[kt][r] = v;
        rmax[r] = fmaxf(rmax[r], v);
      }
    }
  }
#pragma unroll
  for (int m = 1; m <= 8; m <<= 1)
#pragma unroll
    for (int r = 0; r < 8; r++) rmax[r] = fmaxf(rmax[r], __shfl_xor(rmax[r], m, 32));

  float rsum[8];
#pragma unroll
  for (int r = 0; r < 8; r++) rsum[r] = 0.0f;
#pragma unroll
  for (int kt = 0; kt < 16; kt++) {
    if (kt <= mt) {
#pragma unroll
      for (int r = 0; r < 8; r++) {
        float e = __expf(sc[kt][r] - rmax[r]);
        sc[kt][r] = e;
        rsum[r] += e;
      }
    }
  }
#pragma unroll
  for (int m = 1; m <= 8; m <<= 1)
#pragma unroll
    for (int r = 0; r < 8; r++) rsum[r] += __shfl_xor(rsum[r], m, 32);

  // P x V over key pairs (K = 32 per WMMA)
  v8f o[4];
#pragma unroll
  for (int dt = 0; dt < 4; dt++) o[dt] = zero8();
#pragma unroll
  for (int p = 0; p < 8; p++) {
    if (2 * p <= mt) {
      // write P (two 16x16 tiles) C-layout -> LDS -> read as A-fragment
#pragma unroll
      for (int j = 0; j < 2; j++)
#pragma unroll
        for (int r = 0; r < 8; r++) {
          float v = (2 * p + j <= mt) ? sc[2 * p + j][r] : 0.0f;
          pbuf[(r + rofs) * 32 + j * 16 + col] = (bf16_t)v;
        }
      __syncthreads();
      v16bf pa = load_a_bf16(pbuf + (size_t)col * 32 + ((lane >> 4) << 4));
      __syncthreads();
#pragma unroll
      for (int dt = 0; dt < 4; dt++) {
        const bf16_t* vrow = vt + (size_t)(dt * 16 + col) * TT + p * 32 + ((lane >> 4) << 4);
        v16bf vb = *(const v16bf*)vrow;
        o[dt] = WMMA_BF16(pa, vb, o[dt]);
      }
    }
  }

  // normalize + store (concat heads: [B, T, H*D] bf16)
  float rinv[8];
#pragma unroll
  for (int r = 0; r < 8; r++) rinv[r] = 1.0f / rsum[r];
#pragma unroll
  for (int dt = 0; dt < 4; dt++)
#pragma unroll
    for (int r = 0; r < 8; r++)
      obuf[(r + rofs) * 64 + dt * 16 + col] = (bf16_t)(o[dt][r] * rinv[r]);
  __syncthreads();
  int b = bh / HH, h = bh % HH;
  int row = lane >> 1, cb = (lane & 1) * 32;
  bf16_t* dst = aO + ((size_t)(b * TT + m0 + row)) * CC + h * DD + cb;
  const bf16_t* src = obuf + row * 64 + cb;
#pragma unroll
  for (int i = 0; i < 4; i++) *(v8bf*)(dst + i * 8) = *(const v8bf*)(src + i * 8);
}

// ---------------- kernel 4: output projection + bias ----------------
// grid.x = 4096 row tiles; 192 threads = 6 waves, wave w handles n0 = w*64.
__global__ __launch_bounds__(192) void proj_gemm(
    const bf16_t* __restrict__ aO, const bf16_t* __restrict__ packedP,
    const float* __restrict__ bias, float* __restrict__ out) {
  __shared__ bf16_t ab[16 * CC];         // 12 KB staged attention tile
  __shared__ float obuf[6 * 16 * 64];    // 24 KB f32 staging
  int tid = threadIdx.x;
  int m0 = blockIdx.x * 16;

  for (int c = tid; c < (16 * CC) / 8; c += 192)
    *(v8bf*)(ab + c * 8) = *(const v8bf*)(aO + (size_t)m0 * CC + c * 8);
  __syncthreads();

  int wave = tid >> 5, lane = tid & 31;
  int n0 = wave * 64;
  int col = lane & 15, rofs = (lane >> 4) << 3;
  const bf16_t* arow = ab + (size_t)col * CC + ((lane >> 4) << 4);

  v8f acc[4];
#pragma unroll
  for (int nt = 0; nt < 4; nt++) acc[nt] = zero8();
#pragma unroll
  for (int kt = 0; kt < 12; kt++) {
    v16bf a = load_a_bf16(arow + kt * 32);
#pragma unroll
    for (int nt = 0; nt < 4; nt++) {
      v16bf bfrag = *(const v16bf*)(packedP + (size_t)(kt * 24 + wave * 4 + nt) * 512 + lane * 16);
      acc[nt] = WMMA_BF16(a, bfrag, acc[nt]);
    }
  }

  float* ob = obuf + wave * 1024;
#pragma unroll
  for (int nt = 0; nt < 4; nt++) {
    float bv = bias[n0 + nt * 16 + col];
#pragma unroll
    for (int r = 0; r < 8; r++)
      ob[(r + rofs) * 64 + nt * 16 + col] = acc[nt][r] + bv;
  }
  __syncthreads();
  const float* obr = obuf + wave * 1024;
  int row = lane >> 1, cb = (lane & 1) * 32;
  float* dst = out + (size_t)(m0 + row) * CC + n0 + cb;
  const float* src = obr + row * 64 + cb;
#pragma unroll
  for (int i = 0; i < 8; i++) *(v4f*)(dst + i * 4) = *(const v4f*)(src + i * 4);
}

// ---------------- host launcher ----------------
extern "C" void kernel_launch(void* const* d_in, const int* in_sizes, int n_in,
                              void* d_out, int out_size, void* d_ws, size_t ws_size,
                              hipStream_t stream) {
  const float* x     = (const float*)d_in[0];
  const float* Wq    = (const float*)d_in[1];
  const float* Wk    = (const float*)d_in[2];
  const float* Wv    = (const float*)d_in[3];
  const float* Wproj = (const float*)d_in[4];
  const float* bproj = (const float*)d_in[5];
  float* out = (float*)d_out;

  // workspace layout (bf16 elements)
  bf16_t* ws = (bf16_t*)d_ws;
  const size_t PQKV_N = 864 * 512;             // 442368
  const size_t PP_N   = 288 * 512;             // 147456
  const size_t QKV_N  = (size_t)BB * HH * TT * DD;  // 25165824 each
  bf16_t* packedQKV = ws;
  bf16_t* packedP   = packedQKV + PQKV_N;
  bf16_t* Qw  = packedP + PP_N;
  bf16_t* Kw  = Qw + QKV_N;
  bf16_t* Vt  = Kw + QKV_N;
  bf16_t* aO  = Vt + QKV_N;

  pack_weights<<<1152, 32, 0, stream>>>(Wq, Wk, Wv, Wproj, packedQKV, packedP);
  qkv_gemm<<<dim3((BB * TT) / 16), 576, 0, stream>>>(x, packedQKV, Qw, Kw, Vt);
  attention<<<dim3(BB * HH * (TT / 16)), 32, 0, stream>>>(Qw, Kw, Vt, aO);
  proj_gemm<<<dim3((BB * TT) / 16, 1), 192, 0, stream>>>(aO, packedP, bproj, out);
}
